// Transformer_47605417509021
// MI455X (gfx1250) — compile-verified, matrix operand order checked by reading, not run on hardware
//
#include <hip/hip_runtime.h>
#include <stdint.h>

// MoE FFN for MI455X (gfx1250, wave32).
// Routed top-2 implementation: ~103 GFLOP, 402 MB weight traffic (read once).
// Matrix math on v_wmma_f32_16x16x32_f16, weights staged through LDS,
// two M-tiles per wave per iteration (B fragments shared across wmma chains).
// ffn1 LDS layout interleaves w1/w3 16-row groups so all 8 B-fragment
// ds_loads use 16-bit immediate offsets (0/2048/4096/6144 +32) from one base.

#define TTOK 2048   // B*S
#define DDIM 1024
#define IDIM 4096
#define NEXP 8

typedef __attribute__((ext_vector_type(16))) _Float16 v16h;
typedef __attribute__((ext_vector_type(8)))  float    v8f;

union FragU { uint4 q[2]; v16h v; };
union H4   { _Float16 h[4]; uint2 u; };
static_assert(sizeof(v16h) == 32, "v16h must be 32B");

__device__ __forceinline__ float fast_rcp(float x) { return __builtin_amdgcn_rcpf(x); }

// ---------------------------------------------------------------- utilities
__global__ void xconv_kernel(const float* __restrict__ x, _Float16* __restrict__ xh, int n4) {
    int i = blockIdx.x * blockDim.x + threadIdx.x;
    if (i < n4) {
        float4 f = *(const float4*)(x + (size_t)i * 4);
        H4 h; h.h[0] = (_Float16)f.x; h.h[1] = (_Float16)f.y;
              h.h[2] = (_Float16)f.z; h.h[3] = (_Float16)f.w;
        *(uint2*)(xh + (size_t)i * 4) = h.u;
    }
}

// ------------------------------------------------------------------- gating
__global__ void gate_kernel(const float* __restrict__ x, const float* __restrict__ gw,
                            int* __restrict__ tidx, float* __restrict__ tw) {
    int t = blockIdx.x * blockDim.x + threadIdx.x;
    if (t >= TTOK) return;
    float s[NEXP];
#pragma unroll
    for (int e = 0; e < NEXP; ++e) s[e] = 0.f;
    const float* xr = x + (size_t)t * DDIM;
    for (int k = 0; k < DDIM; ++k) {
        float xv = xr[k];
#pragma unroll
        for (int e = 0; e < NEXP; ++e) s[e] += xv * gw[e * DDIM + k];
    }
    // top-2 on scores (softmax is monotone); ties -> lower index (matches top_k)
    int i0 = 0;
#pragma unroll
    for (int e = 1; e < NEXP; ++e) if (s[e] > s[i0]) i0 = e;
    int i1 = (i0 == 0) ? 1 : 0;
#pragma unroll
    for (int e = 0; e < NEXP; ++e) if (e != i0 && s[e] > s[i1]) i1 = e;
    float p1 = __expf(s[i1] - s[i0]);    // p0 = 1
    float inv = fast_rcp(1.f + p1);
    tidx[t * 2 + 0] = i0;  tidx[t * 2 + 1] = i1;
    tw[t * 2 + 0] = inv;  tw[t * 2 + 1] = p1 * inv;
}

__global__ void zero_counts_kernel(int* counts) {
    if (threadIdx.x < NEXP) counts[threadIdx.x] = 0;
}

__global__ void build_lists_kernel(const int* __restrict__ tidx, int* counts,
                                   int* __restrict__ lists) {
    int t = blockIdx.x * blockDim.x + threadIdx.x;
    if (t >= TTOK) return;
#pragma unroll
    for (int a = 0; a < 2; ++a) {
        int e = tidx[t * 2 + a];
        int pos = atomicAdd(&counts[e], 1);
        lists[e * (2 * TTOK) + pos] = t * 2 + a;   // entry id = t*2+a
    }
}

// ---------------------------------------------------- pass A: h = silu(xW1^T)*(xW3^T)
// grid = NEXP * (IDIM/32); block = 128 (4 waves); 2 M-tiles per wave iteration
__global__ void __launch_bounds__(128)
ffn1_kernel(const _Float16* __restrict__ xh,
            const float* __restrict__ w1, const float* __restrict__ w3,
            const int* __restrict__ counts, const int* __restrict__ lists,
            _Float16* __restrict__ hbuf) {
    extern __shared__ char smem_raw[];
    // 16 row-groups; group rg = [w1 row rg | w1 row rg+16 | w3 row rg | w3 row rg+16]
    const int GSTR = 4 * DDIM + 8;                   // 2052 dwords % 64 banks = 4: no conflicts
    _Float16* sw = (_Float16*)smem_raw;              // [16][GSTR]

    int e  = blockIdx.x >> 7;        // IDIM/32 = 128 chunks
    int nc = blockIdx.x & 127;
    int nbase = nc * 32;

    // stage weight chunk (read from HBM exactly once per chunk), float4-wide
    for (int i = threadIdx.x; i < 32 * (DDIM / 4); i += 128) {
        int r = i >> 8;                  // DDIM/4 = 256 vec4 per row
        int c = (i & 255) * 4;
        size_t g = ((size_t)e * IDIM + nbase + r) * DDIM + c;
        float4 f1 = *(const float4*)(w1 + g);
        float4 f3 = *(const float4*)(w3 + g);
        H4 h1; h1.h[0] = (_Float16)f1.x; h1.h[1] = (_Float16)f1.y;
               h1.h[2] = (_Float16)f1.z; h1.h[3] = (_Float16)f1.w;
        H4 h3; h3.h[0] = (_Float16)f3.x; h3.h[1] = (_Float16)f3.y;
               h3.h[2] = (_Float16)f3.z; h3.h[3] = (_Float16)f3.w;
        _Float16* base = sw + (r & 15) * GSTR + (r >> 4) * DDIM;
        *(uint2*)(base + c)            = h1.u;       // w1 slots 0,1
        *(uint2*)(base + 2 * DDIM + c) = h3.u;       // w3 slots 2,3
    }
    __syncthreads();

    int cnt  = counts[e];
    int nmt  = (cnt + 15) >> 4;
    int wave = threadIdx.x >> 5, lane = threadIdx.x & 31;
    int lrow = lane & 15, half = lane >> 4;
    const int* mylist = lists + e * (2 * TTOK);

    for (int mtb = wave * 2; mtb < nmt; mtb += 8) {
        int mt1   = mtb + 1;
        bool has1 = (mt1 < nmt);

        int r0 = mtb * 16 + lrow;
        int ent0 = (r0 < cnt) ? mylist[r0] : 0;
        int r1 = mt1 * 16 + lrow;
        int ent1 = (has1 && r1 < cnt) ? mylist[r1] : 0;

        const _Float16* pa0 = xh + (size_t)(ent0 >> 1) * DDIM + 8 * half;
        const _Float16* pa1 = xh + (size_t)(ent1 >> 1) * DDIM + 8 * half;
        const _Float16* pw  = sw + lrow * GSTR + 8 * half;   // one LDS base, small imms

        v8f acc1[2][2], acc3[2][2];                  // [m-tile][n-tile]
#pragma unroll
        for (int m2 = 0; m2 < 2; ++m2)
#pragma unroll
            for (int nt = 0; nt < 2; ++nt) { acc1[m2][nt] = (v8f)0.f; acc3[m2][nt] = (v8f)0.f; }

        // one k-step per iteration: B frags stay just-in-time ds_loads, no spills
#pragma clang loop unroll(disable)
        for (int kb = 0; kb < DDIM; kb += 32) {
            FragU a0;  a0.q[0]  = *(const uint4*)(pa0);
                       a0.q[1]  = *(const uint4*)(pa0 + 16);
            FragU a1;  a1.q[0]  = *(const uint4*)(pa1);
                       a1.q[1]  = *(const uint4*)(pa1 + 16);
            FragU b10; b10.q[0] = *(const uint4*)(pw);
                       b10.q[1] = *(const uint4*)(pw + 16);
            FragU b11; b11.q[0] = *(const uint4*)(pw + DDIM);
                       b11.q[1] = *(const uint4*)(pw + DDIM + 16);
            FragU b30; b30.q[0] = *(const uint4*)(pw + 2 * DDIM);
                       b30.q[1] = *(const uint4*)(pw + 2 * DDIM + 16);
            FragU b31; b31.q[0] = *(const uint4*)(pw + 3 * DDIM);
                       b31.q[1] = *(const uint4*)(pw + 3 * DDIM + 16);

            acc1[0][0] = __builtin_amdgcn_wmma_f32_16x16x32_f16(
                false, a0.v, false, b10.v, (short)0, acc1[0][0], false, false);
            acc1[1][0] = __builtin_amdgcn_wmma_f32_16x16x32_f16(
                false, a1.v, false, b10.v, (short)0, acc1[1][0], false, false);
            acc1[0][1] = __builtin_amdgcn_wmma_f32_16x16x32_f16(
                false, a0.v, false, b11.v, (short)0, acc1[0][1], false, false);
            acc1[1][1] = __builtin_amdgcn_wmma_f32_16x16x32_f16(
                false, a1.v, false, b11.v, (short)0, acc1[1][1], false, false);
            acc3[0][0] = __builtin_amdgcn_wmma_f32_16x16x32_f16(
                false, a0.v, false, b30.v, (short)0, acc3[0][0], false, false);
            acc3[1][0] = __builtin_amdgcn_wmma_f32_16x16x32_f16(
                false, a1.v, false, b30.v, (short)0, acc3[1][0], false, false);
            acc3[0][1] = __builtin_amdgcn_wmma_f32_16x16x32_f16(
                false, a0.v, false, b31.v, (short)0, acc3[0][1], false, false);
            acc3[1][1] = __builtin_amdgcn_wmma_f32_16x16x32_f16(
                false, a1.v, false, b31.v, (short)0, acc3[1][1], false, false);

            pa0 += 32; pa1 += 32; pw += 32;
        }
        // epilogue: D layout col = lrow, row(vgpr r) = r + 8*half
#pragma unroll
        for (int m2 = 0; m2 < 2; ++m2) {
            if (m2 == 1 && !has1) break;
            int mtx = mtb + m2;
#pragma unroll
            for (int r = 0; r < 8; ++r) {
                int row = mtx * 16 + r + 8 * half;
                if (row < cnt) {
                    int e2 = mylist[row];
#pragma unroll
                    for (int nt = 0; nt < 2; ++nt) {
                        float a1v = acc1[m2][nt][r], a3v = acc3[m2][nt][r];
                        float hv = a1v * fast_rcp(1.f + __expf(-a1v)) * a3v;  // silu*gate
                        hbuf[(size_t)e2 * IDIM + nbase + nt * 16 + lrow] = (_Float16)hv;
                    }
                }
            }
        }
    }
}

// ---------------------------------------------------- pass B: pair = h @ W2^T
// grid = NEXP * (DDIM/16); block = 128 (4 waves); two M-tiles per wave iteration
__global__ void __launch_bounds__(128)
ffn2_kernel(const _Float16* __restrict__ hbuf, const float* __restrict__ w2,
            const int* __restrict__ counts, const int* __restrict__ lists,
            float* __restrict__ pair_out) {
    extern __shared__ char smem_raw[];
    const int LSTR = IDIM + 8;
    _Float16* sw = (_Float16*)smem_raw;              // [16][LSTR]

    int e  = blockIdx.x >> 6;        // DDIM/16 = 64 chunks
    int nc = blockIdx.x & 63;
    int nbase = nc * 16;

    for (int i = threadIdx.x; i < 16 * (IDIM / 4); i += 128) {
        int r = i >> 10;                 // IDIM/4 = 1024 vec4 per row
        int c = (i & 1023) * 4;
        float4 f = *(const float4*)(w2 + ((size_t)e * DDIM + nbase + r) * IDIM + c);
        H4 h; h.h[0] = (_Float16)f.x; h.h[1] = (_Float16)f.y;
              h.h[2] = (_Float16)f.z; h.h[3] = (_Float16)f.w;
        *(uint2*)(sw + r * LSTR + c) = h.u;
    }
    __syncthreads();

    int cnt  = counts[e];
    int nmt  = (cnt + 15) >> 4;
    int wave = threadIdx.x >> 5, lane = threadIdx.x & 31;
    int lrow = lane & 15, half = lane >> 4;
    const int* mylist = lists + e * (2 * TTOK);

    for (int mtb = wave * 2; mtb < nmt; mtb += 8) {
        int mt1   = mtb + 1;
        bool has1 = (mt1 < nmt);

        int row0 = mtb * 16 + lrow;
        int ent0 = (row0 < cnt) ? mylist[row0] : 0;
        int row1 = mt1 * 16 + lrow;
        int ent1 = (has1 && row1 < cnt) ? mylist[row1] : 0;

        const _Float16* pa0 = hbuf + (size_t)ent0 * IDIM + 8 * half;
        const _Float16* pa1 = hbuf + (size_t)ent1 * IDIM + 8 * half;
        const _Float16* pb  = sw + lrow * LSTR + 8 * half;

        v8f acc0 = (v8f)0.f, acc1 = (v8f)0.f;   // two independent wmma chains
#pragma clang loop unroll(disable)
        for (int kb = 0; kb < IDIM; kb += 32) {
            FragU b;  b.q[0]  = *(const uint4*)(pb);
                      b.q[1]  = *(const uint4*)(pb + 16);
            FragU a0; a0.q[0] = *(const uint4*)(pa0);
                      a0.q[1] = *(const uint4*)(pa0 + 16);
            FragU a1; a1.q[0] = *(const uint4*)(pa1);
                      a1.q[1] = *(const uint4*)(pa1 + 16);
            acc0 = __builtin_amdgcn_wmma_f32_16x16x32_f16(
                false, a0.v, false, b.v, (short)0, acc0, false, false);
            acc1 = __builtin_amdgcn_wmma_f32_16x16x32_f16(
                false, a1.v, false, b.v, (short)0, acc1, false, false);
            pa0 += 32; pa1 += 32; pb += 32;
        }
#pragma unroll
        for (int r = 0; r < 8; ++r) {
            int m = r + 8 * half;
            int g0 = mtb * 16 + m;
            if (g0 < cnt) {
                int e2 = mylist[g0];
                pair_out[(size_t)e2 * DDIM + nbase + lrow] = acc0[r];
            }
            int g1 = mt1 * 16 + m;
            if (has1 && g1 < cnt) {
                int e2 = mylist[g1];
                pair_out[(size_t)e2 * DDIM + nbase + lrow] = acc1[r];
            }
        }
    }
}

// ------------------------------------------------------------------ combine
__global__ void combine_kernel(const float* __restrict__ pair_out,
                               const float* __restrict__ tw, float* __restrict__ out) {
    int i = blockIdx.x * blockDim.x + threadIdx.x;
    if (i >= TTOK * DDIM) return;
    int t = i >> 10, d = i & (DDIM - 1);
    float w0 = tw[t * 2 + 0], w1 = tw[t * 2 + 1];
    out[i] = w0 * pair_out[(size_t)(t * 2 + 0) * DDIM + d]
           + w1 * pair_out[(size_t)(t * 2 + 1) * DDIM + d];
}

// ------------------------------------------------------------------- launch
extern "C" void kernel_launch(void* const* d_in, const int* in_sizes, int n_in,
                              void* d_out, int out_size, void* d_ws, size_t ws_size,
                              hipStream_t stream) {
    (void)in_sizes; (void)n_in; (void)out_size; (void)ws_size;
    const float* x  = (const float*)d_in[0];
    const float* gw = (const float*)d_in[1];
    const float* w1 = (const float*)d_in[2];
    const float* w2 = (const float*)d_in[3];
    const float* w3 = (const float*)d_in[4];
    float* out = (float*)d_out;

    char* ws = (char*)d_ws;
    size_t off = 0;
    _Float16* xh   = (_Float16*)(ws + off); off += (size_t)TTOK * DDIM * 2;      // 4 MB
    int*      tidx = (int*)     (ws + off); off += (size_t)TTOK * 2 * 4;         // 16 KB
    float*    tw   = (float*)   (ws + off); off += (size_t)TTOK * 2 * 4;         // 16 KB
    int*      cnts = (int*)     (ws + off); off += 256;                          // counts
    int*      lsts = (int*)     (ws + off); off += (size_t)NEXP * 2 * TTOK * 4;  // 128 KB
    _Float16* hbuf = (_Float16*)(ws + off); off += (size_t)2 * TTOK * IDIM * 2;  // 32 MB
    float*    pair = (float*)   (ws + off); off += (size_t)2 * TTOK * DDIM * 4;  // 16 MB

    int nXD = TTOK * DDIM;
    xconv_kernel<<<(nXD / 4 + 255) / 256, 256, 0, stream>>>(x, xh, nXD / 4);
    gate_kernel<<<(TTOK + 255) / 256, 256, 0, stream>>>(x, gw, tidx, tw);
    zero_counts_kernel<<<1, 32, 0, stream>>>(cnts);
    build_lists_kernel<<<(TTOK + 255) / 256, 256, 0, stream>>>(tidx, cnts, lsts);

    size_t smem1 = (size_t)16 * (4 * DDIM + 8) * sizeof(_Float16);   // ~128 KB
    ffn1_kernel<<<NEXP * (IDIM / 32), 128, smem1, stream>>>(xh, w1, w3, cnts, lsts, hbuf);

    size_t smem2 = (size_t)16 * (IDIM + 8) * sizeof(_Float16);       // ~128 KB
    ffn2_kernel<<<NEXP * (DDIM / 16), 128, smem2, stream>>>(hbuf, w2, cnts, lsts, pair);

    combine_kernel<<<(nXD + 255) / 256, 256, 0, stream>>>(pair, tw, out);
}
